// MultiHeadAttention_51728586113359
// MI455X (gfx1250) — matrix operand order repaired and checked by reading, not compiled, not run on hardware
//
#include <hip/hip_runtime.h>
#include <cstdint>

#ifndef __has_builtin
#define __has_builtin(x) 0
#endif

#if defined(__HIP_DEVICE_COMPILE__) && __has_builtin(__builtin_amdgcn_global_load_async_to_lds_b128)
#define ASYNC_COPY 1
#else
#define ASYNC_COPY 0
#endif

// ---------------- problem constants ----------------
constexpr int DMODEL = 1024;
constexpr int NH     = 16;
constexpr int HD     = 64;      // DK == DV == 64
constexpr int NB     = 4;
constexpr int SEQ    = 2048;
constexpr int MROWS  = NB * SEQ;   // 8192
constexpr int NTK    = SEQ / 64;   // 32 key tiles per row

typedef __attribute__((ext_vector_type(16))) __bf16 v16bf;
typedef __attribute__((ext_vector_type(8)))  float  v8f;
typedef int v4i __attribute__((vector_size(16)));

// ---------------- small device helpers ----------------
__device__ inline v8f zero8() {
  v8f z;
#pragma unroll
  for (int e = 0; e < 8; ++e) z[e] = 0.0f;
  return z;
}

__device__ inline v8f wmma_bf16(v16bf a, v16bf b, v8f c) {
  return __builtin_amdgcn_wmma_f32_16x16x32_bf16(false, a, false, b, (short)0, c,
                                                 false, false);
}

// 16-byte global -> LDS copy (async on CDNA5 path, sync fallback otherwise)
__device__ inline void cp16(__bf16* l, const __bf16* g) {
#if ASYNC_COPY
  __builtin_amdgcn_global_load_async_to_lds_b128(
      (__attribute__((address_space(1))) v4i*)g,
      (__attribute__((address_space(3))) v4i*)l, 0, 0);
#else
  *reinterpret_cast<float4*>(l) = *reinterpret_cast<const float4*>(g);
#endif
}

__device__ inline void wait_async() {
#if ASYNC_COPY
#if __has_builtin(__builtin_amdgcn_s_wait_asynccnt)
  __builtin_amdgcn_s_wait_asynccnt(0);
#else
  asm volatile("s_wait_asynccnt 0x0" ::: "memory");
#endif
#endif
}

// A fragment, 16x32 bf16, from row-major LDS tile A[row][k], stride ld elems.
// lane L: row = L&15, g = L>>4; VGPR v holds K pair k,k+1 with
// k = g*16 + 2v   (contiguous K order, matches load_B/load_BT)
__device__ inline v16bf load_A(const __bf16* a, int ld) {
  const int lane = threadIdx.x & 31;
  const __bf16* r = a + (lane & 15) * ld + (lane >> 4) * 16;
  v16bf out;
#pragma unroll
  for (int v = 0; v < 8; ++v) {
    const int k = 2 * v;
    out[2 * v]     = r[k];
    out[2 * v + 1] = r[k + 1];
  }
  return out;
}

// B fragment, 32x16 bf16, from row-major LDS tile B[k][n], stride ld elems.
// lane L: n = L&15, g = L>>4; VGPR v holds K pair g*16+2v, g*16+2v+1.
__device__ inline v16bf load_B(const __bf16* b, int ld) {
  const int lane = threadIdx.x & 31;
  const int n = lane & 15, g = lane >> 4;
  v16bf out;
#pragma unroll
  for (int v = 0; v < 8; ++v) {
    const int k = g * 16 + 2 * v;
    out[2 * v]     = b[(size_t)k * ld + n];
    out[2 * v + 1] = b[(size_t)(k + 1) * ld + n];
  }
  return out;
}

// B fragment of a TRANSPOSED operand: B[k][n] = t[n*ld + k] (t is N x K
// row-major, e.g. K-cache rows = keys). Per lane this is one contiguous
// 32-byte LDS read.
__device__ inline v16bf load_BT(const __bf16* t, int ld) {
  const int lane = threadIdx.x & 31;
  const __bf16* r = t + (size_t)(lane & 15) * ld + (lane >> 4) * 16;
  return *reinterpret_cast<const v16bf*>(r);
}

// ---------------- kernels ----------------

// fp32 -> bf16, 4 elems / thread
__global__ __launch_bounds__(256) void cvt_bf16(const float* __restrict__ in,
                                                __bf16* __restrict__ out) {
  const int i = (blockIdx.x * 256 + threadIdx.x) * 4;
  const float4 f = *reinterpret_cast<const float4*>(in + i);
  out[i]     = (__bf16)f.x;
  out[i + 1] = (__bf16)f.y;
  out[i + 2] = (__bf16)f.z;
  out[i + 3] = (__bf16)f.w;
}

// 64x64-tile bf16 GEMM, K = 1024, 4 waves / block, WMMA 16x16x32.
// MODE 0: epilogue = +bias, RoPE, store bf16 to [B,H,S,64]
// MODE 1: epilogue = +bias,        store bf16 to [B,H,S,64]
// MODE 2: epilogue = +bias,        store fp32 to [M, DMODEL]
template <int MODE>
__global__ __launch_bounds__(128) void gemm64(
    const __bf16* __restrict__ X, const __bf16* __restrict__ W,
    const float* __restrict__ bias, const float* __restrict__ cosT,
    const float* __restrict__ sinT, __bf16* __restrict__ dstB,
    float* __restrict__ dstF) {
  __shared__ __attribute__((aligned(32))) __bf16 sA[64 * 32];
  __shared__ __attribute__((aligned(32))) __bf16 sB[32 * 64];
  const int tid = threadIdx.x;
  const int w = tid >> 5, lane = tid & 31;
  const int n = lane & 15, g = lane >> 4;
  const int m0 = blockIdx.x * 64;
  const int n0 = blockIdx.y * 64;

  v8f acc[4];
#pragma unroll
  for (int nt = 0; nt < 4; ++nt) acc[nt] = zero8();

  for (int kk = 0; kk < DMODEL; kk += 32) {
    __syncthreads();
    // A tile: 64x32 (256 16B chunks), B tile: 32x64 (256 16B chunks)
    {
      int c = tid;
#pragma unroll
      for (int t = 0; t < 2; ++t, c += 128) {
        const int row = c >> 2, co = (c & 3) * 8;
        cp16(sA + row * 32 + co, X + (size_t)(m0 + row) * DMODEL + kk + co);
      }
      c = tid;
#pragma unroll
      for (int t = 0; t < 2; ++t, c += 128) {
        const int row = c >> 3, co = (c & 7) * 8;
        cp16(sB + row * 64 + co, W + (size_t)(kk + row) * DMODEL + n0 + co);
      }
    }
    wait_async();
    __syncthreads();
    const v16bf a = load_A(sA + w * 16 * 32, 32);
#pragma unroll
    for (int nt = 0; nt < 4; ++nt) {
      const v16bf b = load_B(sB + nt * 16, 64);
      acc[nt] = wmma_bf16(a, b, acc[nt]);
    }
  }

  // ---- epilogue ----
  if constexpr (MODE == 2) {
#pragma unroll
    for (int v = 0; v < 8; ++v) {
      const int row = m0 + w * 16 + g * 8 + v;
#pragma unroll
      for (int nt = 0; nt < 4; ++nt) {
        const int col = n0 + nt * 16 + n;
        dstF[(size_t)row * DMODEL + col] = acc[nt][v] + bias[col];
      }
    }
  } else {
    const int h = blockIdx.y;  // N-tile == one head (64 cols)
#pragma unroll
    for (int v = 0; v < 8; ++v) {
      const int row = m0 + w * 16 + g * 8 + v;
      const int b = row >> 11, s = row & (SEQ - 1);
      const size_t base = ((size_t)(b * NH + h) * SEQ + s) * HD;
      if constexpr (MODE == 1) {
#pragma unroll
        for (int nt = 0; nt < 4; ++nt) {
          const int d = nt * 16 + n;
          dstB[base + d] = (__bf16)(acc[nt][v] + bias[h * HD + d]);
        }
      } else {  // MODE 0: RoPE, pairs (i, i+32) -> (2i, 2i+1)
#pragma unroll
        for (int nt = 0; nt < 2; ++nt) {
          const int ii = nt * 16 + n;  // 0..31
          const float x1 = acc[nt][v] + bias[h * HD + ii];
          const float x2 = acc[nt + 2][v] + bias[h * HD + ii + 32];
          const float c  = cosT[s * (HD / 2) + ii];
          const float sn = sinT[s * (HD / 2) + ii];
          dstB[base + 2 * ii]     = (__bf16)(x1 * c - x2 * sn);
          dstB[base + 2 * ii + 1] = (__bf16)(x1 * sn + x2 * c);
        }
      }
    }
  }
}

// Causal attention. grid = (S/64, H, B), 128 threads (4 waves).
// Wave w owns q-rows [qt*64 + w*16, +16). Two passes:
//  pass 1: online (max,sum) stats over the causal range
//  pass 2: recompute scores, write normalized probs to attn, accumulate P@V
__global__ __launch_bounds__(128) void attn_kernel(
    const __bf16* __restrict__ Qr, const __bf16* __restrict__ Kr,
    const __bf16* __restrict__ Vr, float* __restrict__ attnO,
    __bf16* __restrict__ ctx) {
  __shared__ __attribute__((aligned(32))) __bf16 sQ[64 * 64];
  __shared__ __attribute__((aligned(32))) __bf16 sK[64 * 64];
  __shared__ __attribute__((aligned(32))) __bf16 sV[64 * 64];
  __shared__ __attribute__((aligned(32))) __bf16 sP[64 * 64];

  const int tid = threadIdx.x, w = tid >> 5, lane = tid & 31;
  const int n = lane & 15, g = lane >> 4;
  const int i = blockIdx.x, h = blockIdx.y, b = blockIdx.z;
  const size_t bh = (size_t)(b * NH + h);
  const size_t headbase = bh * SEQ * HD;

  // Q tile -> LDS (8 KB), then per-wave A fragments (persist in VGPRs)
  {
    const __bf16* qg = Qr + headbase + (size_t)i * 64 * HD;
    for (int c = tid; c < 512; c += 128) cp16(sQ + c * 8, qg + c * 8);
    wait_async();
  }
  __syncthreads();
  const v16bf aQ0 = load_A(sQ + w * 16 * 64, 64);
  const v16bf aQ1 = load_A(sQ + w * 16 * 64 + 32, 64);

  float m[8], l[8];
#pragma unroll
  for (int v = 0; v < 8; ++v) { m[v] = -1e30f; l[v] = 0.0f; }
  const int qbase = i * 64 + w * 16 + g * 8;

  // ---------------- pass 1: softmax statistics ----------------
  for (int j = 0; j <= i; ++j) {
    __syncthreads();
    const __bf16* kg = Kr + headbase + (size_t)j * 64 * HD;
    for (int c = tid; c < 512; c += 128) cp16(sK + c * 8, kg + c * 8);
    wait_async();
    __syncthreads();

    v8f sc[4];
#pragma unroll
    for (int nt = 0; nt < 4; ++nt) sc[nt] = zero8();
#pragma unroll
    for (int ks = 0; ks < 2; ++ks) {
      const v16bf a = ks ? aQ1 : aQ0;
#pragma unroll
      for (int nt = 0; nt < 4; ++nt) {
        const v16bf bk = load_BT(sK + nt * 16 * 64 + ks * 32, 64);
        sc[nt] = wmma_bf16(a, bk, sc[nt]);
      }
    }
#pragma unroll
    for (int v = 0; v < 8; ++v) {
      const int q = qbase + v;
      float mx = -1e30f;
#pragma unroll
      for (int nt = 0; nt < 4; ++nt) {
        const int k = j * 64 + nt * 16 + n;
        float s = sc[nt][v] * 0.125f;  // 1/sqrt(64)
        if (k > q) s = -1e30f;         // causal
        sc[nt][v] = s;
        mx = fmaxf(mx, s);
      }
#pragma unroll
      for (int msk = 1; msk < 16; msk <<= 1) mx = fmaxf(mx, __shfl_xor(mx, msk));
      const float mn = fmaxf(m[v], mx);
      float rs = 0.0f;
#pragma unroll
      for (int nt = 0; nt < 4; ++nt) rs += __expf(sc[nt][v] - mn);
#pragma unroll
      for (int msk = 1; msk < 16; msk <<= 1) rs += __shfl_xor(rs, msk);
      l[v] = l[v] * __expf(m[v] - mn) + rs;
      m[v] = mn;
    }
  }
#pragma unroll
  for (int v = 0; v < 8; ++v) l[v] = 1.0f / l[v];

  // ---------------- pass 2: write attn, accumulate O = P@V ----------------
  v8f o[4];
#pragma unroll
  for (int nt = 0; nt < 4; ++nt) o[nt] = zero8();

  for (int j = 0; j < NTK; ++j) {
    if (j <= i) {
      __syncthreads();
      const __bf16* kg = Kr + headbase + (size_t)j * 64 * HD;
      const __bf16* vg = Vr + headbase + (size_t)j * 64 * HD;
      for (int c = tid; c < 512; c += 128) {
        cp16(sK + c * 8, kg + c * 8);
        cp16(sV + c * 8, vg + c * 8);
      }
      wait_async();
      __syncthreads();

      v8f sc[4];
#pragma unroll
      for (int nt = 0; nt < 4; ++nt) sc[nt] = zero8();
#pragma unroll
      for (int ks = 0; ks < 2; ++ks) {
        const v16bf a = ks ? aQ1 : aQ0;
#pragma unroll
        for (int nt = 0; nt < 4; ++nt) {
          const v16bf bk = load_BT(sK + nt * 16 * 64 + ks * 32, 64);
          sc[nt] = wmma_bf16(a, bk, sc[nt]);
        }
      }
#pragma unroll
      for (int v = 0; v < 8; ++v) {
        const int q = qbase + v;
        float* arow = attnO + (bh * SEQ + q) * (size_t)SEQ + (size_t)j * 64;
#pragma unroll
        for (int nt = 0; nt < 4; ++nt) {
          const int k = j * 64 + nt * 16 + n;
          const float s = sc[nt][v] * 0.125f;
          const float p = (k > q) ? 0.0f : __expf(s - m[v]) * l[v];
          arow[nt * 16 + n] = p;
          sP[(w * 16 + g * 8 + v) * 64 + nt * 16 + n] = (__bf16)p;
        }
      }
      __syncthreads();  // sP transpose staging visible
      const v16bf aP0 = load_A(sP + w * 16 * 64, 64);
      const v16bf aP1 = load_A(sP + w * 16 * 64 + 32, 64);
#pragma unroll
      for (int ks = 0; ks < 2; ++ks) {
        const v16bf a = ks ? aP1 : aP0;
#pragma unroll
        for (int nt = 0; nt < 4; ++nt) {
          const v16bf bv = load_B(sV + ks * 32 * 64 + nt * 16, 64);
          o[nt] = wmma_bf16(a, bv, o[nt]);
        }
      }
    } else {
      // fully-masked tile: softmax is exactly 0 there; just store zeros
#pragma unroll
      for (int v = 0; v < 8; ++v) {
        const int q = qbase + v;
        float* arow = attnO + (bh * SEQ + q) * (size_t)SEQ + (size_t)j * 64;
#pragma unroll
        for (int nt = 0; nt < 4; ++nt) arow[nt * 16 + n] = 0.0f;
      }
    }
  }

  // context out: [b, s, h*64 + d] bf16 (probabilities already normalized)
#pragma unroll
  for (int v = 0; v < 8; ++v) {
    const int row = b * SEQ + i * 64 + w * 16 + g * 8 + v;
#pragma unroll
    for (int nt = 0; nt < 4; ++nt)
      ctx[(size_t)row * DMODEL + h * HD + nt * 16 + n] = (__bf16)o[nt][v];
  }
}

// ---------------- host launcher ----------------
extern "C" void kernel_launch(void* const* d_in, const int* in_sizes, int n_in,
                              void* d_out, int out_size, void* d_ws,
                              size_t ws_size, hipStream_t stream) {
  (void)in_sizes; (void)n_in; (void)out_size; (void)ws_size;

  const float* query = (const float*)d_in[0];
  const float* key_i = (const float*)d_in[1];
  const float* value = (const float*)d_in[2];
  const float* Wq = (const float*)d_in[3];
  const float* bq = (const float*)d_in[4];
  const float* Wk = (const float*)d_in[5];
  const float* bk = (const float*)d_in[6];
  const float* Wv = (const float*)d_in[7];
  const float* bv = (const float*)d_in[8];
  const float* Wo = (const float*)d_in[9];
  const float* bo = (const float*)d_in[10];
  const float* rc = (const float*)d_in[11];
  const float* rs = (const float*)d_in[12];
  // d_in[13] = mask (bool) -- causality implemented directly

  char* ws = (char*)d_ws;
  size_t off = 0;
  auto salloc = [&](size_t bytes) {
    char* p = ws + off;
    off += (bytes + 255) & ~(size_t)255;
    return p;
  };
  const size_t nBig = (size_t)MROWS * DMODEL;     // 8,388,608
  const size_t nW   = (size_t)DMODEL * DMODEL;    // 1,048,576
  __bf16* Xq  = (__bf16*)salloc(nBig * 2);
  __bf16* Xk  = (__bf16*)salloc(nBig * 2);
  __bf16* Xv  = (__bf16*)salloc(nBig * 2);
  __bf16* Wqb = (__bf16*)salloc(nW * 2);
  __bf16* Wkb = (__bf16*)salloc(nW * 2);
  __bf16* Wvb = (__bf16*)salloc(nW * 2);
  __bf16* Wob = (__bf16*)salloc(nW * 2);
  __bf16* Qr  = (__bf16*)salloc(nBig * 2);        // [B,H,S,64] roped
  __bf16* Kr  = (__bf16*)salloc(nBig * 2);
  __bf16* Vr  = (__bf16*)salloc(nBig * 2);
  __bf16* ctx = (__bf16*)salloc(nBig * 2);        // [B,S,H*64]

  // fp32 -> bf16 conversions
  cvt_bf16<<<(int)(nBig / 1024), 256, 0, stream>>>(query, Xq);
  cvt_bf16<<<(int)(nBig / 1024), 256, 0, stream>>>(key_i, Xk);
  cvt_bf16<<<(int)(nBig / 1024), 256, 0, stream>>>(value, Xv);
  cvt_bf16<<<(int)(nW / 1024), 256, 0, stream>>>(Wq, Wqb);
  cvt_bf16<<<(int)(nW / 1024), 256, 0, stream>>>(Wk, Wkb);
  cvt_bf16<<<(int)(nW / 1024), 256, 0, stream>>>(Wv, Wvb);
  cvt_bf16<<<(int)(nW / 1024), 256, 0, stream>>>(Wo, Wob);

  // projections (+RoPE for Q,K), WMMA bf16
  const dim3 ggrid(MROWS / 64, DMODEL / 64);
  gemm64<0><<<ggrid, 128, 0, stream>>>(Xq, Wqb, bq, rc, rs, Qr, nullptr);
  gemm64<0><<<ggrid, 128, 0, stream>>>(Xk, Wkb, bk, rc, rs, Kr, nullptr);
  gemm64<1><<<ggrid, 128, 0, stream>>>(Xv, Wvb, bv, nullptr, nullptr, Vr, nullptr);

  // attention: output layout = (output fp32 [B,S,D], attn fp32 [B,H,S,S])
  float* out0  = (float*)d_out;
  float* attnP = out0 + (size_t)MROWS * DMODEL;
  attn_kernel<<<dim3(NTK, NH, NB), 128, 0, stream>>>(Qr, Kr, Vr, attnP, ctx);

  // output projection + bias -> fp32
  gemm64<2><<<ggrid, 128, 0, stream>>>(ctx, Wob, bo, nullptr, nullptr, nullptr, out0);
}
